// __TemporalDecoderLayer_4715874091531
// MI455X (gfx1250) — compile-verified, matrix operand order checked by reading, not run on hardware
//
#include <hip/hip_runtime.h>
#include <hip/hip_bf16.h>
#include <math.h>

// ---------------------------------------------------------------------------
// TemporalDecoderLayer for MI455X (gfx1250, wave32, WMMA).
// Mixed precision: bf16 WMMA (v_wmma_f32_16x16x32_bf16) with f32 accumulate;
// LayerNorm / softmax / residuals in f32.  GEMM staging uses CDNA5 async
// global->LDS copies (ASYNCcnt) with double-buffered LDS tiles.
// ---------------------------------------------------------------------------

typedef unsigned short u16;
typedef __attribute__((ext_vector_type(16))) __bf16 bf16x16;
typedef __attribute__((ext_vector_type(8)))  float  f32x8;
typedef __attribute__((ext_vector_type(4)))  int    i32x4;

#ifndef __has_builtin
#define __has_builtin(x) 0
#endif
#if __has_builtin(__builtin_amdgcn_global_load_async_to_lds_b128)
#define HAVE_ASYNC_LDS 1
#else
#define HAVE_ASYNC_LDS 0
#endif

#if HAVE_ASYNC_LDS
#define AS1 __attribute__((address_space(1)))
#define AS3 __attribute__((address_space(3)))
__device__ __forceinline__ void async_g2l_b128(const u16* g, u16* l) {
  // GLOBAL_LOAD_ASYNC_TO_LDS_B128: 16B memory->LDS per lane, no VGPR round
  // trip, tracked by ASYNCcnt.  Builtin signature (from clang diagnostics):
  // (v4i addrspace(1)*, v4i addrspace(3)*, imm offset, imm cpol).
  __builtin_amdgcn_global_load_async_to_lds_b128(
      (AS1 i32x4*)g, (AS3 i32x4*)l, 0, 0);
}
#if __has_builtin(__builtin_amdgcn_s_wait_asynccnt)
#define WAIT_ASYNC0() __builtin_amdgcn_s_wait_asynccnt(0)
#else
#define WAIT_ASYNC0() asm volatile("s_wait_asynccnt 0x0" ::: "memory")
#endif
#endif  // HAVE_ASYNC_LDS

#define BATCH 4
#define SEQ   1024
#define NBLK  4
#define DM    1024
#define NH    16
#define DHD   64
#define DFFN  4096
#define TOK   (BATCH*SEQ)        // 4096 query tokens
#define MROW  (BATCH*SEQ*NBLK)   // 16384 memory rows

// GEMM tiling: block = 256 threads = 8 waves, tile 128x128, K-step 32.
// Each wave owns a 32x64 sub-tile = 2x4 WMMA 16x16 tiles.
#define BM 128
#define BN 128
#define BK 32
#define KP 40   // padded LDS row stride (elems) to spread banks

union FragB { bf16x16 v; unsigned u[8]; };

__device__ __forceinline__ u16 f2b(float f) {
  unsigned u = __float_as_uint(f);
  u += 0x7FFFu + ((u >> 16) & 1u);          // round-to-nearest-even
  return (u16)(u >> 16);
}
__device__ __forceinline__ float b2f(u16 h) {
  return __uint_as_float(((unsigned)h) << 16);
}

// ---------------------------------------------------------------------------
__global__ void cvt_bf16_kernel(const float* __restrict__ in,
                                u16* __restrict__ out, int n) {
  int i = blockIdx.x * 256 + threadIdx.x;
  if (i < n) out[i] = f2b(in[i]);
}

// ---------------------------------------------------------------------------
// Row LayerNorm over D=1024, optional f32 and bf16 outputs.
__global__ void ln_kernel(const float* __restrict__ x,
                          const float* __restrict__ w,
                          const float* __restrict__ b,
                          float* __restrict__ outf,
                          u16*   __restrict__ outb) {
  const int row = blockIdx.x;
  const int tid = threadIdx.x;
  const float* xr = x + (size_t)row * DM;
  __shared__ float red[256];

  float s = 0.f;
  for (int i = tid; i < DM; i += 256) s += xr[i];
  red[tid] = s; __syncthreads();
  for (int off = 128; off > 0; off >>= 1) {
    if (tid < off) red[tid] += red[tid + off];
    __syncthreads();
  }
  const float mu = red[0] * (1.0f / DM);
  __syncthreads();

  float v = 0.f;
  for (int i = tid; i < DM; i += 256) { float d = xr[i] - mu; v += d * d; }
  red[tid] = v; __syncthreads();
  for (int off = 128; off > 0; off >>= 1) {
    if (tid < off) red[tid] += red[tid + off];
    __syncthreads();
  }
  const float rstd = rsqrtf(red[0] * (1.0f / DM) + 1e-5f);

  for (int i = tid; i < DM; i += 256) {
    float val = (xr[i] - mu) * rstd * w[i] + b[i];
    if (outf) outf[(size_t)row * DM + i] = val;
    if (outb) outb[(size_t)row * DM + i] = f2b(val);
  }
}

// ---------------------------------------------------------------------------
// Core WMMA GEMM: C[M,N] = alpha * A[M,K](bf16) @ W[N,K](bf16)^T  (+bias,
// +f32 residual, optional relu; f32 and/or bf16 stores).  Both A rows and W
// rows are K-contiguous, which matches the ISA A / B fragment layouts.
// Double-buffered LDS + async global->LDS staging: tile i+1 is in flight
// while tile i feeds the WMMAs; one s_wait_asynccnt + one barrier per step.
__device__ __forceinline__ void gemm_core(
    const u16* __restrict__ A, int lda,
    const u16* __restrict__ Bw, int ldb,
    const float* __restrict__ bias,
    const float* __restrict__ resid,
    float* __restrict__ Cf, u16* __restrict__ Cb, int ldc,
    int K, float alpha, int relu) {
  __shared__ __align__(16) u16 As[2][BM * KP];
  __shared__ __align__(16) u16 Bs[2][BN * KP];
  const int tid  = threadIdx.x;
  const int lane = tid & 31;
  const int wave = tid >> 5;
  const int wm = wave >> 1;   // 0..3  (row group of 32)
  const int wn = wave & 1;    // 0..1  (col group of 64)
  const int m0 = blockIdx.y * BM;
  const int n0 = blockIdx.x * BN;

  f32x8 acc[2][4];
  const f32x8 zf = {0.f,0.f,0.f,0.f,0.f,0.f,0.f,0.f};
#pragma unroll
  for (int t = 0; t < 2; t++)
#pragma unroll
    for (int u2 = 0; u2 < 4; u2++) acc[t][u2] = zf;

  const int lrow = tid >> 2;        // 0..63
  const int lcg  = (tid & 3) * 8;   // 0,8,16,24

  auto issue = [&](int buf, int k0) {
#pragma unroll
    for (int p = 0; p < 2; p++) {
      const int row = lrow + p * 64;
#if HAVE_ASYNC_LDS
      async_g2l_b128(&A [(size_t)(m0 + row) * lda + k0 + lcg], &As[buf][row * KP + lcg]);
      async_g2l_b128(&Bw[(size_t)(n0 + row) * ldb + k0 + lcg], &Bs[buf][row * KP + lcg]);
#else
      *(uint4*)&As[buf][row * KP + lcg] =
          *(const uint4*)&A[(size_t)(m0 + row) * lda + k0 + lcg];
      *(uint4*)&Bs[buf][row * KP + lcg] =
          *(const uint4*)&Bw[(size_t)(n0 + row) * ldb + k0 + lcg];
#endif
    }
  };

  const int nk = K / BK;
  issue(0, 0);
#if HAVE_ASYNC_LDS
  WAIT_ASYNC0();
#endif
  __syncthreads();

  for (int i = 0; i < nk; i++) {
    const int cur = i & 1;
    if (i + 1 < nk) issue(1 - cur, (i + 1) * BK);

    const int kh = lane >> 4;   // which K half-group this lane holds
    const int ml = lane & 15;   // M (for A) / N (for B) index within tile
    FragB af[2], bfr[4];
#pragma unroll
    for (int t = 0; t < 2; t++) {
      const int row = wm * 32 + t * 16 + ml;
#pragma unroll
      for (int v = 0; v < 8; v++) {
        const int k = (v >> 2) * 16 + kh * 8 + (v & 3) * 2;
        af[t].u[v] = *(const unsigned*)&As[cur][row * KP + k];
      }
    }
#pragma unroll
    for (int u2 = 0; u2 < 4; u2++) {
      const int row = wn * 64 + u2 * 16 + ml;
#pragma unroll
      for (int v = 0; v < 8; v++) {
        const int k = (v >> 2) * 16 + kh * 8 + (v & 3) * 2;
        bfr[u2].u[v] = *(const unsigned*)&Bs[cur][row * KP + k];
      }
    }
#pragma unroll
    for (int t = 0; t < 2; t++)
#pragma unroll
      for (int u2 = 0; u2 < 4; u2++)
        acc[t][u2] = __builtin_amdgcn_wmma_f32_16x16x32_bf16(
            false, af[t].v, false, bfr[u2].v, (short)0, acc[t][u2],
            false, false);

    if (i + 1 < nk) {
#if HAVE_ASYNC_LDS
      WAIT_ASYNC0();
#endif
      __syncthreads();
    }
  }

  // Epilogue: C/D layout -> lane%16 = N, M = vgpr + 8*(lane/16)
  const int en = lane & 15;
  const int emh = (lane >> 4) * 8;
#pragma unroll
  for (int t = 0; t < 2; t++) {
#pragma unroll
    for (int u2 = 0; u2 < 4; u2++) {
      const int n = n0 + wn * 64 + u2 * 16 + en;
      const float bv = bias ? bias[n] : 0.0f;
#pragma unroll
      for (int v = 0; v < 8; v++) {
        const int m = m0 + wm * 32 + t * 16 + emh + v;
        float val = acc[t][u2][v] * alpha + bv;
        if (resid) val += resid[(size_t)m * ldc + n];
        if (relu)  val = val > 0.f ? val : 0.f;
        if (Cf) Cf[(size_t)m * ldc + n] = val;
        if (Cb) Cb[(size_t)m * ldc + n] = f2b(val);
      }
    }
  }
}

__global__ void gemm_kernel(const u16* __restrict__ A, int lda,
                            const u16* __restrict__ Bw, int ldb,
                            const float* __restrict__ bias,
                            const float* __restrict__ resid,
                            float* __restrict__ Cf, u16* __restrict__ Cb,
                            int ldc, int K, float alpha, int relu) {
  gemm_core(A, lda, Bw, ldb, bias, resid, Cf, Cb, ldc, K, alpha, relu);
}

// Self-attention scores: for batch z = b*NH+h, S x S = Q @ K^T * 1/sqrt(DH).
// Q rows and K rows are both DH-contiguous inside the packed qkv buffer.
__global__ void attn_scores_kernel(const u16* __restrict__ qkv,
                                   u16* __restrict__ probs) {
  const int z = blockIdx.z, b = z >> 4, h = z & 15;
  const u16* Aq = qkv + (size_t)b * SEQ * 3 * DM + h * DHD;
  const u16* Bk = qkv + (size_t)b * SEQ * 3 * DM + DM + h * DHD;
  u16* C = probs + (size_t)z * SEQ * SEQ;
  gemm_core(Aq, 3 * DM, Bk, 3 * DM, nullptr, nullptr, nullptr, C, SEQ,
            DHD, 0.125f /* 1/sqrt(64) */, 0);
}

// ---------------------------------------------------------------------------
// Masked softmax over keys for one (b, q) row across all 16 heads; writes
// normalized probs (bf16, in place) and the head-averaged attention output.
__global__ void self_softmax_kernel(u16* __restrict__ probs,
                                    const int* __restrict__ pmask,
                                    float* __restrict__ attn_avg) {
  const int r = blockIdx.x, b = r >> 10, q = r & 1023;
  const int tid = threadIdx.x;
  __shared__ float red[256];

  float mk[4];
#pragma unroll
  for (int j = 0; j < 4; j++) {
    const int i = tid + j * 256;
    mk[j] = (pmask[b * SEQ + i] == 1) ? 0.0f : -INFINITY;
  }
  float avg[4] = {0.f, 0.f, 0.f, 0.f};

  for (int h = 0; h < NH; h++) {
    u16* pr = probs + ((size_t)(b * NH + h) * SEQ + q) * SEQ;
    float rv[4];
    float m = -INFINITY;
#pragma unroll
    for (int j = 0; j < 4; j++) {
      rv[j] = b2f(pr[tid + j * 256]) + mk[j];
      m = fmaxf(m, rv[j]);
    }
    red[tid] = m; __syncthreads();
    for (int off = 128; off > 0; off >>= 1) {
      if (tid < off) red[tid] = fmaxf(red[tid], red[tid + off]);
      __syncthreads();
    }
    m = red[0]; __syncthreads();

    float s = 0.f;
#pragma unroll
    for (int j = 0; j < 4; j++) { rv[j] = __expf(rv[j] - m); s += rv[j]; }
    red[tid] = s; __syncthreads();
    for (int off = 128; off > 0; off >>= 1) {
      if (tid < off) red[tid] += red[tid + off];
      __syncthreads();
    }
    s = red[0]; __syncthreads();

    const float inv = 1.0f / s;
#pragma unroll
    for (int j = 0; j < 4; j++) {
      const float p = rv[j] * inv;
      pr[tid + j * 256] = f2b(p);
      avg[j] += p * (1.0f / NH);
    }
  }
#pragma unroll
  for (int j = 0; j < 4; j++)
    attn_avg[(size_t)r * SEQ + tid + j * 256] = avg[j];
}

// ---------------------------------------------------------------------------
// A*V: per (b,h), O[S,DH] = P[S,S](bf16) @ V[S,DH](bf16).  V is K-major in
// memory (key rows, DH contiguous), so the B tile is staged [k][n] in LDS and
// repacked into the mirrored WMMA B fragment (lane = N, K pairs per VGPR).
// Same async double-buffered staging as gemm_core.
__global__ void attn_av_kernel(const u16* __restrict__ qkv,
                               const u16* __restrict__ probs,
                               u16* __restrict__ o) {
  const int z = blockIdx.z, b = z >> 4, h = z & 15;
  const u16* P = probs + (size_t)z * SEQ * SEQ;
  const u16* V = qkv + (size_t)b * SEQ * 3 * DM + 2 * DM + h * DHD;
  u16* C = o + (size_t)b * SEQ * DM + h * DHD;
  __shared__ __align__(16) u16 As[2][BM * KP];
  __shared__ __align__(16) u16 Bs[2][BK * 72];   // [k][n], padded n-stride 72
  const int tid = threadIdx.x, lane = tid & 31, wave = tid >> 5;
  const int wm = wave >> 1, wn = wave & 1;       // wave tile: 32 x 32
  const int m0 = blockIdx.y * BM;

  f32x8 acc[2][2];
  const f32x8 zf = {0.f,0.f,0.f,0.f,0.f,0.f,0.f,0.f};
#pragma unroll
  for (int t = 0; t < 2; t++)
#pragma unroll
    for (int u2 = 0; u2 < 2; u2++) acc[t][u2] = zf;

  const int lrow = tid >> 2, lcg = (tid & 3) * 8;
  const int bkk = tid >> 3, bng = (tid & 7) * 8;

  auto issue = [&](int buf, int k0) {
#pragma unroll
    for (int p = 0; p < 2; p++) {
      const int row = lrow + p * 64;
#if HAVE_ASYNC_LDS
      async_g2l_b128(&P[(size_t)(m0 + row) * SEQ + k0 + lcg], &As[buf][row * KP + lcg]);
#else
      *(uint4*)&As[buf][row * KP + lcg] =
          *(const uint4*)&P[(size_t)(m0 + row) * SEQ + k0 + lcg];
#endif
    }
#if HAVE_ASYNC_LDS
    async_g2l_b128(&V[(size_t)(k0 + bkk) * (3 * DM) + bng], &Bs[buf][bkk * 72 + bng]);
#else
    *(uint4*)&Bs[buf][bkk * 72 + bng] =
        *(const uint4*)&V[(size_t)(k0 + bkk) * (3 * DM) + bng];
#endif
  };

  const int nk = SEQ / BK;
  issue(0, 0);
#if HAVE_ASYNC_LDS
  WAIT_ASYNC0();
#endif
  __syncthreads();

  for (int i = 0; i < nk; i++) {
    const int cur = i & 1;
    if (i + 1 < nk) issue(1 - cur, (i + 1) * BK);

    const int kh = lane >> 4, ml = lane & 15;
    FragB af[2], bfr[2];
#pragma unroll
    for (int t = 0; t < 2; t++) {
      const int row = wm * 32 + t * 16 + ml;
#pragma unroll
      for (int v = 0; v < 8; v++) {
        const int k = (v >> 2) * 16 + kh * 8 + (v & 3) * 2;
        af[t].u[v] = *(const unsigned*)&As[cur][row * KP + k];
      }
    }
#pragma unroll
    for (int u2 = 0; u2 < 2; u2++) {
      const int col = wn * 32 + u2 * 16 + ml;
#pragma unroll
      for (int v = 0; v < 8; v++) {
        const int kb = (v >> 2) * 16 + kh * 8 + (v & 3) * 2;
        const unsigned lo = Bs[cur][kb * 72 + col];
        const unsigned hi = Bs[cur][(kb + 1) * 72 + col];
        bfr[u2].u[v] = lo | (hi << 16);
      }
    }
#pragma unroll
    for (int t = 0; t < 2; t++)
#pragma unroll
      for (int u2 = 0; u2 < 2; u2++)
        acc[t][u2] = __builtin_amdgcn_wmma_f32_16x16x32_bf16(
            false, af[t].v, false, bfr[u2].v, (short)0, acc[t][u2],
            false, false);

    if (i + 1 < nk) {
#if HAVE_ASYNC_LDS
      WAIT_ASYNC0();
#endif
      __syncthreads();
    }
  }

  const int en = lane & 15, emh = (lane >> 4) * 8;
#pragma unroll
  for (int t = 0; t < 2; t++)
#pragma unroll
    for (int u2 = 0; u2 < 2; u2++) {
      const int n = wn * 32 + u2 * 16 + en;
#pragma unroll
      for (int v = 0; v < 8; v++) {
        const int m = m0 + wm * 32 + t * 16 + emh + v;
        C[(size_t)m * DM + n] = f2b(acc[t][u2][v]);
      }
    }
}

// ---------------------------------------------------------------------------
// Block cross-attention (NB=4 is too small for WMMA): one block per token.
// scores[h][n] = q . k_n over DH, softmax over n, co = sum_n w*v_n,
// x = tgt_n + co.  Also writes cross_attn_w output.
__global__ void cross_attn_kernel(const u16* __restrict__ q,
                                  const u16* __restrict__ k,
                                  const u16* __restrict__ v,
                                  const float* __restrict__ tgt_n,
                                  float* __restrict__ x,
                                  float* __restrict__ out_cross) {
  const int r = blockIdx.x;           // b*SEQ + s
  const int b = r >> 10, s = r & 1023;
  const u16* qr = q + (size_t)r * DM;
  const u16* kr = k + (size_t)r * NBLK * DM;
  const u16* vr = v + (size_t)r * NBLK * DM;
  const int tid = threadIdx.x;
  __shared__ float ps[256];
  __shared__ float sc[64];   // [h*4+n]
  __shared__ float wt[64];

  // 64 dot products of length 64, 4 threads each.
  const int p = tid >> 2, sub = tid & 3;
  const int h = p >> 2, n = p & 3;
  float a = 0.f;
#pragma unroll
  for (int j = 0; j < 16; j++) {
    const int d = h * DHD + sub * 16 + j;
    a += b2f(qr[d]) * b2f(kr[n * DM + d]);
  }
  ps[tid] = a; __syncthreads();
  if (tid < 64)
    sc[tid] = (ps[tid*4] + ps[tid*4+1] + ps[tid*4+2] + ps[tid*4+3]) * 0.125f;
  __syncthreads();

  if (tid < 16) {
    float m = -INFINITY;
#pragma unroll
    for (int nn = 0; nn < 4; nn++) m = fmaxf(m, sc[tid*4+nn]);
    float e[4], sum = 0.f;
#pragma unroll
    for (int nn = 0; nn < 4; nn++) { e[nn] = __expf(sc[tid*4+nn] - m); sum += e[nn]; }
#pragma unroll
    for (int nn = 0; nn < 4; nn++) wt[tid*4+nn] = e[nn] / sum;
  }
  __syncthreads();

  if (tid < 64) {
    const int hh = tid >> 2, nn = tid & 3;
    out_cross[((size_t)(b * NH + hh) * SEQ + s) * NBLK + nn] = wt[tid];
  }

#pragma unroll
  for (int j = 0; j < 4; j++) {
    const int d = tid + j * 256;
    const int hh = d >> 6;
    float co = 0.f;
#pragma unroll
    for (int nn = 0; nn < 4; nn++) co += wt[hh*4+nn] * b2f(vr[nn * DM + d]);
    x[(size_t)r * DM + d] = tgt_n[(size_t)r * DM + d] + co;
  }
}

// ---------------------------------------------------------------------------
extern "C" void kernel_launch(void* const* d_in, const int* in_sizes, int n_in,
                              void* d_out, int out_size, void* d_ws, size_t ws_size,
                              hipStream_t stream) {
  (void)in_sizes; (void)n_in; (void)out_size; (void)ws_size;
  const float* tgt       = (const float*)d_in[0];
  const float* memory    = (const float*)d_in[1];
  const int*   pmask     = (const int*)  d_in[2];
  const float* ln_tgt_w  = (const float*)d_in[3];
  const float* ln_tgt_b  = (const float*)d_in[4];
  const float* ln_mem_w  = (const float*)d_in[5];
  const float* ln_mem_b  = (const float*)d_in[6];
  const float* ln1_w     = (const float*)d_in[7];
  const float* ln1_b     = (const float*)d_in[8];
  const float* ln2_w     = (const float*)d_in[9];
  const float* ln2_b     = (const float*)d_in[10];
  const float* Wq        = (const float*)d_in[11];
  const float* bq        = (const float*)d_in[12];
  const float* Wk        = (const float*)d_in[13];
  const float* bk        = (const float*)d_in[14];
  const float* Wv        = (const float*)d_in[15];
  const float* bv        = (const float*)d_in[16];
  const float* in_proj_w = (const float*)d_in[17];
  const float* in_proj_b = (const float*)d_in[18];
  const float* out_proj_w= (const float*)d_in[19];
  const float* out_proj_b= (const float*)d_in[20];
  const float* W1        = (const float*)d_in[21];
  const float* b1        = (const float*)d_in[22];
  const float* W2        = (const float*)d_in[23];
  const float* b2        = (const float*)d_in[24];

  float* out_ff    = (float*)d_out;                          // [B,S,D]
  float* out_attn  = out_ff   + (size_t)TOK * SEQ;           // [B,S,S]
  float* out_cross = out_attn + (size_t)TOK * SEQ;           // [B,H,S,1,NB]

  // --- workspace carve (bump allocator, 256B aligned) ---
  char* wsp = (char*)d_ws;
  auto take = [&](size_t elems, size_t esz) -> void* {
    void* p = (void*)wsp;
    wsp += ((elems * esz + 255) & ~(size_t)255);
    return p;
  };
  float* tgt_n_f = (float*)take((size_t)TOK * DM, 4);
  float* x_f     = (float*)take((size_t)TOK * DM, 4);
  float* x1_f    = (float*)take((size_t)TOK * DM, 4);
  float* y_f     = (float*)take((size_t)TOK * DM, 4);
  float* y2_f    = (float*)take((size_t)TOK * DM, 4);
  u16* tgt_n_b = (u16*)take((size_t)TOK * DM, 2);
  u16* mem_n_b = (u16*)take((size_t)MROW * DM, 2);
  u16* q_b     = (u16*)take((size_t)TOK * DM, 2);
  u16* k_b     = (u16*)take((size_t)MROW * DM, 2);
  u16* v_b     = (u16*)take((size_t)MROW * DM, 2);
  u16* x1_b    = (u16*)take((size_t)TOK * DM, 2);
  u16* qkv_b   = (u16*)take((size_t)TOK * 3 * DM, 2);
  u16* probs_b = (u16*)take((size_t)BATCH * NH * SEQ * SEQ, 2);
  u16* o_b     = (u16*)take((size_t)TOK * DM, 2);
  u16* y2_b    = (u16*)take((size_t)TOK * DM, 2);
  u16* h1_b    = (u16*)take((size_t)TOK * DFFN, 2);
  u16* wq_b    = (u16*)take((size_t)DM * DM, 2);
  u16* wk_b    = (u16*)take((size_t)DM * DM, 2);
  u16* wv_b    = (u16*)take((size_t)DM * DM, 2);
  u16* winp_b  = (u16*)take((size_t)3 * DM * DM, 2);
  u16* wout_b  = (u16*)take((size_t)DM * DM, 2);
  u16* w1_b    = (u16*)take((size_t)DFFN * DM, 2);
  u16* w2_b    = (u16*)take((size_t)DM * DFFN, 2);

  // --- weight conversions to bf16 ---
  struct { const float* s; u16* d; int n; } cv[7] = {
    {Wq, wq_b, DM*DM}, {Wk, wk_b, DM*DM}, {Wv, wv_b, DM*DM},
    {in_proj_w, winp_b, 3*DM*DM}, {out_proj_w, wout_b, DM*DM},
    {W1, w1_b, DFFN*DM}, {W2, w2_b, DM*DFFN},
  };
  for (int i = 0; i < 7; i++)
    cvt_bf16_kernel<<<(cv[i].n + 255)/256, 256, 0, stream>>>(cv[i].s, cv[i].d, cv[i].n);

  // --- pre-norms ---
  ln_kernel<<<TOK, 256, 0, stream>>>(tgt, ln_tgt_w, ln_tgt_b, tgt_n_f, tgt_n_b);
  ln_kernel<<<MROW, 256, 0, stream>>>(memory, ln_mem_w, ln_mem_b, nullptr, mem_n_b);

  // --- cross-attention projections (WMMA GEMMs) ---
  gemm_kernel<<<dim3(DM/BN, TOK/BM), 256, 0, stream>>>(
      tgt_n_b, DM, wq_b, DM, bq, nullptr, nullptr, q_b, DM, DM, 1.0f, 0);
  gemm_kernel<<<dim3(DM/BN, MROW/BM), 256, 0, stream>>>(
      mem_n_b, DM, wk_b, DM, bk, nullptr, nullptr, k_b, DM, DM, 1.0f, 0);
  gemm_kernel<<<dim3(DM/BN, MROW/BM), 256, 0, stream>>>(
      mem_n_b, DM, wv_b, DM, bv, nullptr, nullptr, v_b, DM, DM, 1.0f, 0);

  // --- fused cross-attention + residual; then LN1 ---
  cross_attn_kernel<<<TOK, 256, 0, stream>>>(q_b, k_b, v_b, tgt_n_f, x_f, out_cross);
  ln_kernel<<<TOK, 256, 0, stream>>>(x_f, ln1_w, ln1_b, x1_f, x1_b);

  // --- self-attention ---
  gemm_kernel<<<dim3(3*DM/BN, TOK/BM), 256, 0, stream>>>(
      x1_b, DM, winp_b, DM, in_proj_b, nullptr, nullptr, qkv_b, 3*DM, DM, 1.0f, 0);
  attn_scores_kernel<<<dim3(SEQ/BN, SEQ/BM, BATCH*NH), 256, 0, stream>>>(qkv_b, probs_b);
  self_softmax_kernel<<<TOK, 256, 0, stream>>>(probs_b, pmask, out_attn);
  attn_av_kernel<<<dim3(1, SEQ/BM, BATCH*NH), 256, 0, stream>>>(qkv_b, probs_b, o_b);
  gemm_kernel<<<dim3(DM/BN, TOK/BM), 256, 0, stream>>>(
      o_b, DM, wout_b, DM, out_proj_b, x1_f, y_f, nullptr, DM, DM, 1.0f, 0);

  // --- LN2 + feed-forward ---
  ln_kernel<<<TOK, 256, 0, stream>>>(y_f, ln2_w, ln2_b, y2_f, y2_b);
  gemm_kernel<<<dim3(DFFN/BN, TOK/BM), 256, 0, stream>>>(
      y2_b, DM, w1_b, DM, b1, nullptr, nullptr, h1_b, DFFN, DM, 1.0f, 1 /*relu*/);
  gemm_kernel<<<dim3(DM/BN, TOK/BM), 256, 0, stream>>>(
      h1_b, DFFN, w2_b, DFFN, b2, y2_f, out_ff, nullptr, DM, DFFN, 1.0f, 0);
}
